// Net_91293824844249
// MI455X (gfx1250) — matrix operand order repaired and checked by reading, not compiled, not run on hardware
//
#include <hip/hip_runtime.h>
#include <cstdint>
#include <cstddef>

// ===========================================================================
// MI455X (gfx1250) implementation.
//  - ConvGRU stack (~93% of FLOPs) -> persistent per-sample workgroup,
//    hidden state in LDS, all channel-convs as im2col GEMMs on
//    v_wmma_f32_16x16x32_f16 (f16 in, f32 accumulate).
//    Double-buffered im2col stage (2x ds_store_b128 per thread per chunk,
//    LDS tap table, one barrier per K-chunk).
//  - NALU featurizer + training-mode BN -> VALU kernels with deterministic
//    two-stage (partial -> fixed-order) batch statistics.
// ===========================================================================

typedef __attribute__((ext_vector_type(16))) _Float16 v16h;
typedef __attribute__((ext_vector_type(8)))  float    v8f;

#define T_SEQ 18
#define BATCH 256
#define NS    (BATCH*T_SEQ)          // 4608 (t,b) samples
#define NPIX  125                    // 25 x 5
#define NP    128                    // padded pixel count (8 x 16-col tiles)
#define HID   48
#define EPS_NALU 1e-7f
#define EPS_BN   1e-5f
#define BSTR  40                     // bstage row stride in halves (16B aligned, conflict-free)
#define NTAP  2466                   // sum of Ktrue over the 6 GRU layers

// ---------------- workspace layout (float units unless noted) --------------
enum : int {
  MU1_OFF = 0,            RS1_OFF = 288,          // [18][16]
  MU2_OFF = 576,          RS2_OFF = 1440,         // [18][48]
  MU3_OFF = 2304,         RS3_OFF = 2340,         // [18][2]
  MU4_OFF = 2376,         RS4_OFF = 2388,         // [12]
  P1S_OFF = 4096,
  P1Q_OFF = P1S_OFF + NS*16,
  P2S_OFF = P1Q_OFF + NS*16,
  P2Q_OFF = P2S_OFF + NS*48,
  P3S_OFF = P2Q_OFF + NS*48,
  P3Q_OFF = P3S_OFF + NS*2,
  P4S_OFF = P3Q_OFF + NS*2,
  P4Q_OFF = P4S_OFF + BATCH*12,
  WM1_OFF = P4Q_OFF + BATCH*12,     // NALU effective weights tanh(W)*sig(M)
  WM2_OFF = WM1_OFF + 135,
  WM3_OFF = WM2_OFF + 4608,
  WM4_OFF = WM3_OFF + 48,
  WM_END  = WM4_OFF + 5184,
  APACK_F = ((WM_END + 7) & ~7),    // f16 GRU weight image starts here (16B aligned)
  APACK_HALVES = 359424,            // 144 rows x (480+96+864+96+864+96)
  X4_F  = APACK_F + APACK_HALVES/2, // conv3 output, [NS][125] f32
  H6_F  = X4_F + NS*NPIX,           // [256][48][125] f32
  O4_F  = H6_F + BATCH*HID*NPIX,    // [256][12][125] f32
};

// per-GRU-layer metadata
__constant__ int c_cx[6]    = {2, 48, 48, 48, 48, 48};
__constant__ int c_ks[6]    = {3, 1, 3, 1, 3, 1};
__constant__ int c_ktrue[6] = {450, 96, 864, 96, 864, 96};
__constant__ int c_kpad[6]  = {480, 96, 864, 96, 864, 96};
__constant__ int c_aoff[6]  = {0, 69120, 82944, 207360, 221184, 345600}; // halves
__constant__ int c_toff[7]  = {0, 450, 546, 1410, 1506, 2370, 2466};     // tap-table bases

struct GruW { const float* w[18]; };                       // (layer,mat) -> weight
struct GruB { const float* br[6]; const float* bu[6]; const float* bc[6]; };

__device__ __forceinline__ float sigm(float x){ return 1.f/(1.f+expf(-x)); }

// ---------------------------------------------------------------------------
// weight packing
// ---------------------------------------------------------------------------
__global__ void k_pack_wm(const float* c1W, const float* c1M,
                          const float* c2W, const float* c2M,
                          const float* c3W, const float* c3M,
                          const float* c4W, const float* c4M,
                          float* wsf)
{
  int i = blockIdx.x*blockDim.x + threadIdx.x;
  const float *W, *M; int off, n;
  if      (i < 135)  { W=c1W; M=c1M; off=WM1_OFF; n=i; }
  else if (i < 4743) { W=c2W; M=c2M; off=WM2_OFF; n=i-135; }
  else if (i < 4791) { W=c3W; M=c3M; off=WM3_OFF; n=i-4743; }
  else if (i < 9975) { W=c4W; M=c4M; off=WM4_OFF; n=i-4791; }
  else return;
  wsf[off + n] = tanhf(W[n]) * sigm(M[n]);
}

// pack GRU conv weights fp32 [48][Cin][kh][kw] -> f16 row-major [48][Kpad],
// zero padded in K.  im2col k index = ci*9+dy*3+dx == natural flatten.
__global__ void k_pack_gru(GruW gw, _Float16* Ap)
{
  int bid = blockIdx.x;                 // 6*3*48 = 864
  int L = bid / 144, rem = bid % 144, mat = rem / 48, row = rem % 48;
  int Kp = c_kpad[L], Kt = c_ktrue[L];
  _Float16* dst = Ap + c_aoff[L] + (mat*48 + row)*Kp;
  const float* src = gw.w[L*3 + mat] + row*Kt;
  for (int k = threadIdx.x; k < Kp; k += blockDim.x)
    dst[k] = (k < Kt) ? (_Float16)src[k] : (_Float16)0.f;
}

// ---------------------------------------------------------------------------
// step1 featurizer (NALU convs + per-timestep BN statistics)
// ---------------------------------------------------------------------------
__device__ void dev_conv1(const float* wm1, const float* G1,
                          const float* xt, const float* lx, float* x1,
                          int tid, int nt)
{
  for (int o = tid; o < 15*NPIX; o += nt){
    int co = o / NPIX, p = o - co*NPIX;
    int y = p/5, xp = p - y*5;
    float a=0.f, m=0.f, g=0.f;
    for (int dy=0; dy<3; ++dy)
      for (int dx=0; dx<3; ++dx){
        int yy=y+dy-1, xx=xp+dx-1;
        if ((unsigned)yy<25u && (unsigned)xx<5u){
          int pp = yy*5+xx;
          float w = wm1[co*9 + dy*3 + dx];
          a += w*xt[pp]; m += w*lx[pp];
          g += G1[co*9 + dy*3 + dx]*xt[pp];
        }
      }
    float gs = sigm(g);
    x1[o] = gs*a + (1.f-gs)*expf(m);
  }
}

__device__ void dev_c1n(const float* wsf, const float* g1, const float* b1, int t,
                        const float* xt, const float* x1, float* c1n, float* lc1,
                        int tid, int nt)
{
  for (int i = tid; i < 16*NPIX; i += nt){
    int c = i / NPIX, p = i - c*NPIX;
    float v = (c==0) ? xt[p] : x1[(c-1)*NPIX + p];
    float vn = (v - wsf[MU1_OFF + t*16 + c])*wsf[RS1_OFF + t*16 + c]*g1[c] + b1[c];
    c1n[i] = vn; lc1[i] = logf(fabsf(vn)+EPS_NALU);
  }
}

__device__ void dev_conv2(const float* wm2, const float* G2,
                          const float* c1n, const float* lc1, float* x2,
                          int tid, int nt)
{
  for (int o = tid; o < 32*NPIX; o += nt){
    int co = o / NPIX, p = o - co*NPIX;
    int y = p/5, xp = p - y*5;
    float a=0.f, m=0.f, g=0.f;
    for (int ci=0; ci<16; ++ci)
      for (int dy=0; dy<3; ++dy)
        for (int dx=0; dx<3; ++dx){
          int yy=y+dy-1, xx=xp+dx-1;
          if ((unsigned)yy<25u && (unsigned)xx<5u){
            int pp = ci*NPIX + yy*5+xx;
            int wi = (co*16+ci)*9 + dy*3 + dx;
            a += wm2[wi]*c1n[pp]; m += wm2[wi]*lc1[pp]; g += G2[wi]*c1n[pp];
          }
        }
    float gs = sigm(g);
    x2[o] = gs*a + (1.f-gs)*expf(m);
  }
}

__global__ void k_bn1_stats(const float* x, float* wsf, const float* G1)
{
  int s = blockIdx.x, tid = threadIdx.x;
  __shared__ float xt[NPIX], lx[NPIX], x1[15*NPIX];
  for (int p = tid; p < NPIX; p += 128){
    float v = x[s*NPIX+p]; xt[p]=v; lx[p]=logf(fabsf(v)+EPS_NALU);
  }
  __syncthreads();
  dev_conv1(wsf+WM1_OFF, G1, xt, lx, x1, tid, 128);
  __syncthreads();
  if (tid < 16){
    float sm=0.f, sq=0.f;
    for (int p=0;p<NPIX;++p){
      float v = (tid==0) ? xt[p] : x1[(tid-1)*NPIX+p];
      sm += v; sq += v*v;
    }
    wsf[P1S_OFF + s*16 + tid] = sm;
    wsf[P1Q_OFF + s*16 + tid] = sq;
  }
}

__global__ void k_bn2_stats(const float* x, float* wsf, const float* G1, const float* G2,
                            const float* g1, const float* b1)
{
  int s = blockIdx.x, t = s % T_SEQ, tid = threadIdx.x;
  __shared__ float xt[NPIX], lx[NPIX], x1[15*NPIX], c1n[16*NPIX], lc1[16*NPIX], x2[32*NPIX];
  for (int p = tid; p < NPIX; p += 128){
    float v = x[s*NPIX+p]; xt[p]=v; lx[p]=logf(fabsf(v)+EPS_NALU);
  }
  __syncthreads();
  dev_conv1(wsf+WM1_OFF, G1, xt, lx, x1, tid, 128);
  __syncthreads();
  dev_c1n(wsf, g1, b1, t, xt, x1, c1n, lc1, tid, 128);
  __syncthreads();
  dev_conv2(wsf+WM2_OFF, G2, c1n, lc1, x2, tid, 128);
  __syncthreads();
  if (tid < 48){
    const float* src = (tid<16) ? (c1n + tid*NPIX) : (x2 + (tid-16)*NPIX);
    float sm=0.f, sq=0.f;
    for (int p=0;p<NPIX;++p){ float v=src[p]; sm+=v; sq+=v*v; }
    wsf[P2S_OFF + s*48 + tid] = sm;
    wsf[P2Q_OFF + s*48 + tid] = sq;
  }
}

__global__ void k_bn3_x4(const float* x, const float* timing, float* wsf,
                         const float* G1, const float* G2, const float* G3,
                         const float* g1, const float* b1,
                         const float* g2, const float* b2)
{
  int s = blockIdx.x, t = s % T_SEQ, tid = threadIdx.x;
  __shared__ float xt[NPIX], lx[NPIX], x1[15*NPIX], c1n[16*NPIX], lc1[16*NPIX], x2[32*NPIX], x4s[NPIX];
  for (int p = tid; p < NPIX; p += 128){
    float v = x[s*NPIX+p]; xt[p]=v; lx[p]=logf(fabsf(v)+EPS_NALU);
  }
  __syncthreads();
  dev_conv1(wsf+WM1_OFF, G1, xt, lx, x1, tid, 128);
  __syncthreads();
  dev_c1n(wsf, g1, b1, t, xt, x1, c1n, lc1, tid, 128);
  __syncthreads();
  dev_conv2(wsf+WM2_OFF, G2, c1n, lc1, x2, tid, 128);
  __syncthreads();
  for (int p = tid; p < NPIX; p += 128){
    float a=0.f, m=0.f, g=0.f;
    for (int c=0;c<48;++c){
      float v  = (c<16) ? c1n[c*NPIX+p] : x2[(c-16)*NPIX+p];
      float vn = (v - wsf[MU2_OFF + t*48 + c])*wsf[RS2_OFF + t*48 + c]*g2[c] + b2[c];
      float w  = wsf[WM3_OFF + c];
      a += w*vn; m += w*logf(fabsf(vn)+EPS_NALU); g += G3[c]*vn;
    }
    float gs = sigm(g);
    float o  = gs*a + (1.f-gs)*expf(m);
    x4s[p] = o;
    wsf[X4_F + s*NPIX + p] = o;
  }
  __syncthreads();
  if (tid == 0){
    float sm=0.f, sq=0.f;
    for (int p=0;p<NPIX;++p){ sm+=x4s[p]; sq+=x4s[p]*x4s[p]; }
    wsf[P3S_OFF + s*2+0]=sm; wsf[P3Q_OFF + s*2+0]=sq;
    float tv = timing[s];                       // s = b*T + t
    wsf[P3S_OFF + s*2+1]=125.f*tv; wsf[P3Q_OFF + s*2+1]=125.f*tv*tv;
  }
}

// deterministic fixed-order reduction of per-sample partials -> mu, rsqrt(var+eps)
__global__ void k_finalize(float* wsf, int psoff, int pqoff, int muoff, int rsoff,
                           int nch, int nt)
{
  int j = blockIdx.x*blockDim.x + threadIdx.x;
  if (j >= nch*nt) return;
  int t = j / nch, c = j - t*nch;
  float sm=0.f, sq=0.f;
  for (int bb=0; bb<BATCH; ++bb){
    int idx = (bb*nt + t)*nch + c;
    sm += wsf[psoff + idx]; sq += wsf[pqoff + idx];
  }
  float cnt = (float)(BATCH*NPIX);
  float mu  = sm/cnt;
  float var = sq/cnt - mu*mu;
  wsf[muoff + t*nch + c] = mu;
  wsf[rsoff + t*nch + c] = rsqrtf(var + EPS_BN);
}

// ---------------------------------------------------------------------------
// WMMA helpers (layouts per CDNA5 ISA 7.12.2)
// ---------------------------------------------------------------------------
__device__ __forceinline__ v8f wmma_f16(v16h a, v16h b, v8f c){
  return __builtin_amdgcn_wmma_f32_16x16x32_f16(false, a, false, b, (short)0, c, false, false);
}

// A (16x32 f16) from row-major [48][Kpad] image.
// lanes 0-15: M=lane,   K in kc*32+0..15
// lanes16-31: M=lane-16, K in kc*32+16..31.
__device__ __forceinline__ v16h load_Afrag(const _Float16* A, int Kp, int mt, int kc, int lane)
{
  int row = mt*16 + (lane & 15);
  int kb  = kc*32 + ((lane & 16) ? 16 : 0);
  const _Float16* p = A + row*Kp + kb;
  union { uint4 q[2]; v16h v; } f;
  f.q[0] = *(const uint4*)(p);
  f.q[1] = *(const uint4*)(p + 8);
  return f.v;
}

// B (32x16 f16) from LDS stage laid out [n][BSTR] (transposed im2col chunk).
// lanes 0-15: N=lane,   K=0..15;  lanes 16-31: N=lane-16, K=16..31.
__device__ __forceinline__ v16h load_Bfrag(const _Float16* bst, int wave, int lane)
{
  int row = wave*16 + (lane & 15);
  int ko  = (lane & 16);
  const _Float16* p = bst + row*BSTR + ko;
  union { uint4 q[2]; v16h v; } f;
  f.q[0] = *(const uint4*)(p);
  f.q[1] = *(const uint4*)(p + 8);
  return f.v;
}

// im2col stage, tap-table driven.  Each thread owns a fixed (n, 16-wide K half)
// pair; builds 16 halves in registers and writes two b128 LDS stores.
// tap[k] = (ci*NP)<<8 | dy<<4 | dx   (dy,dx raw in 0..2; 1x1 convs use dy=dx=1)
__device__ __forceinline__ void stage_B2(_Float16* dst, const _Float16* act,
                                         const uint32_t* tap, int Kt, int kbase,
                                         int n, int kh, int y, int xcol, bool nvalid)
{
  union { _Float16 h[16]; uint4 q[2]; } v;
#pragma unroll
  for (int j=0;j<16;++j){
    int k = kbase + kh + j;
    _Float16 val = (_Float16)0.f;
    if (nvalid && k < Kt){
      uint32_t e = tap[k];
      int yy = y    + (int)((e >> 4) & 15u) - 1;
      int xx = xcol + (int)(e & 15u) - 1;
      if ((unsigned)yy < 25u && (unsigned)xx < 5u)
        val = act[(e >> 8) + yy*5 + xx];
    }
    v.h[j] = val;
  }
  uint4* o = (uint4*)(dst + n*BSTR + kh);
  o[0] = v.q[0]; o[1] = v.q[1];
}

// ---------------------------------------------------------------------------
// persistent ConvGRU scan: 1 block = 1 batch sample, T loop in-kernel,
// hidden state in LDS, every conv = WMMA GEMM, double-buffered im2col stage.
// ---------------------------------------------------------------------------
__global__ void __launch_bounds__(256)
k_gru(float* wsf, const _Float16* __restrict__ Ap,
      const float* __restrict__ timing, const float* g3, const float* b3, GruB gb)
{
  int b = blockIdx.x, tid = threadIdx.x, lane = tid & 31, wave = tid >> 5;
  extern __shared__ _Float16 dsm[];
  _Float16* hb   = dsm;                   // [6][48][128]
  _Float16* act  = hb  + 6*HID*NP;        // [96][128] concat(x_in, h) / (x_in, r*h)
  _Float16* ug   = act + 96*NP;           // [48][128] update gate
  _Float16* bst0 = ug  + HID*NP;          // [128][BSTR] im2col buffer 0
  _Float16* bst1 = bst0 + NP*BSTR;        // [128][BSTR] im2col buffer 1
  uint32_t* taps = (uint32_t*)(bst1 + NP*BSTR);  // [NTAP]

  // build tap tables (all layers) + zero hidden state
  for (int i=tid; i<NTAP; i+=256){
    int L = 0;
    while (L < 5 && i >= c_toff[L+1]) ++L;
    int k = i - c_toff[L];
    int ci, dy, dx;
    if (c_ks[L] == 3){ ci = k/9; int r9 = k - ci*9; dy = r9/3; dx = r9 - dy*3; }
    else             { ci = k; dy = 1; dx = 1; }
    taps[i] = ((uint32_t)(ci*NP) << 8) | ((uint32_t)dy << 4) | (uint32_t)dx;
  }
  for (int i=tid; i<6*HID*NP; i+=256) hb[i] = (_Float16)0.f;
  __syncthreads();

  // per-thread staging invariants
  int  sn  = tid >> 1;               // pixel column n in [0,128)
  int  skh = (tid & 1) << 4;         // K half: 0 or 16
  int  sy  = sn / 5, sx = sn - sy*5;
  bool snv = sn < NPIX;

  const v8f z8 = {0.f,0.f,0.f,0.f,0.f,0.f,0.f,0.f};

  for (int t=0; t<T_SEQ; ++t){
    int s = b*T_SEQ + t;
    float mu0 = wsf[MU3_OFF+t*2+0], rs0 = wsf[RS3_OFF+t*2+0];
    float mu1 = wsf[MU3_OFF+t*2+1], rs1 = wsf[RS3_OFF+t*2+1];
    float tn  = (timing[s] - mu1)*rs1*g3[1] + b3[1];

    for (int L=0; L<6; ++L){
      int cx = c_cx[L], Kt = c_ktrue[L], Kp = c_kpad[L];
      int nch = Kp >> 5;
      const _Float16* Ar = Ap + c_aoff[L];
      const _Float16* Au = Ar + 48*Kp;
      const _Float16* Ac = Au + 48*Kp;
      const uint32_t* tapL = taps + c_toff[L];

      // build concat input: rows [0,cx) = x_in, rows [cx,cx+48) = h_L
      if (L == 0){
        for (int p=tid; p<NP; p+=256){
          float v0=0.f, v1=0.f;
          if (p < NPIX){
            v0 = (wsf[X4_F + s*NPIX + p] - mu0)*rs0*g3[0] + b3[0];
            v1 = tn;
          }
          act[p] = (_Float16)v0; act[NP+p] = (_Float16)v1;
        }
        for (int i=tid; i<HID*NP; i+=256) act[2*NP + i] = hb[i];
      } else {
        for (int i=tid; i<HID*NP; i+=256){
          act[i]          = hb[(L-1)*HID*NP + i];   // x_in = h_{L-1} (this t)
          act[HID*NP + i] = hb[L*HID*NP + i];       // h_L (previous t)
        }
      }
      __syncthreads();

      // ---- r & u gate GEMMs (share the staged B tile) ----
      v8f accR[3] = {z8,z8,z8};
      v8f accU[3] = {z8,z8,z8};
      stage_B2(bst0, act, tapL, Kt, 0, sn, skh, sy, sx, snv);
      __syncthreads();
      for (int kc=0; kc<nch; ++kc){
        _Float16* cur = (kc & 1) ? bst1 : bst0;
        _Float16* nxt = (kc & 1) ? bst0 : bst1;
        v16h bf = load_Bfrag(cur, wave, lane);
        if (kc + 1 < nch)
          stage_B2(nxt, act, tapL, Kt, (kc+1)*32, sn, skh, sy, sx, snv);
#pragma unroll
        for (int mt=0; mt<3; ++mt){
          accR[mt] = wmma_f16(load_Afrag(Ar, Kp, mt, kc, lane), bf, accR[mt]);
          accU[mt] = wmma_f16(load_Afrag(Au, Kp, mt, kc, lane), bf, accU[mt]);
        }
        __syncthreads();
      }
      {
        int col = (wave<<4) + (lane & 15);
        int rb  = (lane & 16) ? 8 : 0;
#pragma unroll
        for (int mt=0; mt<3; ++mt){
#pragma unroll
          for (int v=0; v<8; ++v){
            int row = mt*16 + rb + v;
            float rv = sigm(accR[mt][v] + gb.br[L][row]);
            float uv = sigm(accU[mt][v] + gb.bu[L][row]);
            ug[row*NP + col] = (_Float16)uv;
            float hv = (float)hb[L*HID*NP + row*NP + col];
            act[(cx+row)*NP + col] = (_Float16)(rv*hv);   // candidate input: r*h
          }
        }
      }
      __syncthreads();

      // ---- candidate GEMM ----
      v8f accC[3] = {z8,z8,z8};
      stage_B2(bst0, act, tapL, Kt, 0, sn, skh, sy, sx, snv);
      __syncthreads();
      for (int kc=0; kc<nch; ++kc){
        _Float16* cur = (kc & 1) ? bst1 : bst0;
        _Float16* nxt = (kc & 1) ? bst0 : bst1;
        v16h bf = load_Bfrag(cur, wave, lane);
        if (kc + 1 < nch)
          stage_B2(nxt, act, tapL, Kt, (kc+1)*32, sn, skh, sy, sx, snv);
#pragma unroll
        for (int mt=0; mt<3; ++mt)
          accC[mt] = wmma_f16(load_Afrag(Ac, Kp, mt, kc, lane), bf, accC[mt]);
        __syncthreads();
      }
      {
        int col = (wave<<4) + (lane & 15);
        int rb  = (lane & 16) ? 8 : 0;
#pragma unroll
        for (int mt=0; mt<3; ++mt){
#pragma unroll
          for (int v=0; v<8; ++v){
            int row = mt*16 + rb + v;
            float cv  = tanhf(accC[mt][v] + gb.bc[L][row]);
            float uv  = (float)ug[row*NP + col];
            float hv  = (float)hb[L*HID*NP + row*NP + col];
            float res = (L > 0) ? (float)act[row*NP + col] : 0.f;  // residual = x_in
            hb[L*HID*NP + row*NP + col] = (_Float16)(res + uv*hv + (1.f-uv)*cv);
          }
        }
      }
      __syncthreads();
    }
  }

  for (int i=tid; i<HID*NPIX; i+=256){
    int c = i/NPIX, p = i - c*NPIX;
    wsf[H6_F + b*HID*NPIX + i] = (float)hb[5*HID*NP + c*NP + p];
  }
}

// ---------------------------------------------------------------------------
// conv4 NALU (48->12, 3x3) + bn4 partial stats
// ---------------------------------------------------------------------------
__global__ void k_conv4(float* wsf, const float* G4)
{
  int b = blockIdx.x, tid = threadIdx.x;
  __shared__ float hs[HID*NPIX], lh[HID*NPIX], o4s[12*NPIX];
  for (int i=tid; i<HID*NPIX; i+=128){
    float v = wsf[H6_F + b*HID*NPIX + i];
    hs[i]=v; lh[i]=logf(fabsf(v)+EPS_NALU);
  }
  __syncthreads();
  for (int o=tid; o<12*NPIX; o+=128){
    int co=o/NPIX, p=o-co*NPIX, y=p/5, xp=p-y*5;
    float a=0.f, m=0.f, g=0.f;
    for (int ci=0; ci<48; ++ci)
      for (int dy=0; dy<3; ++dy)
        for (int dx=0; dx<3; ++dx){
          int yy=y+dy-1, xx=xp+dx-1;
          if ((unsigned)yy<25u && (unsigned)xx<5u){
            int pp = ci*NPIX + yy*5+xx;
            int wi = (co*48+ci)*9 + dy*3+dx;
            float w = wsf[WM4_OFF + wi];
            a += w*hs[pp]; m += w*lh[pp]; g += G4[wi]*hs[pp];
          }
        }
    float gs = sigm(g);
    float ov = gs*a + (1.f-gs)*expf(m);
    o4s[o] = ov;
    wsf[O4_F + b*12*NPIX + o] = ov;
  }
  __syncthreads();
  if (tid < 12){
    float sm=0.f, sq=0.f;
    for (int p=0;p<NPIX;++p){ float v=o4s[tid*NPIX+p]; sm+=v; sq+=v*v; }
    wsf[P4S_OFF + b*12 + tid]=sm; wsf[P4Q_OFF + b*12 + tid]=sq;
  }
}

__global__ void k_out(const float* wsf, const float* wout, const float* bout,
                      const float* g4, const float* b4, float* out)
{
  int i = blockIdx.x*256 + threadIdx.x;
  if (i >= BATCH*NPIX) return;
  int b = i / NPIX, p = i - b*NPIX;
  float acc = bout[0];
  for (int c=0;c<12;++c){
    float v  = wsf[O4_F + (b*12 + c)*NPIX + p];
    float vn = (v - wsf[MU4_OFF+c])*wsf[RS4_OFF+c]*g4[c] + b4[c];
    acc += wout[c]*vn;
  }
  out[i] = acc;
}

// ---------------------------------------------------------------------------
extern "C" void kernel_launch(void* const* d_in, const int* in_sizes, int n_in,
                              void* d_out, int out_size, void* d_ws, size_t ws_size,
                              hipStream_t stream)
{
  (void)in_sizes; (void)n_in; (void)out_size; (void)ws_size;

  const float* x      = (const float*)d_in[0];
  const float* timing = (const float*)d_in[1];
  const float* c1W = (const float*)d_in[2],  *c1M = (const float*)d_in[3],  *c1G = (const float*)d_in[4];
  const float* bn1g= (const float*)d_in[5],  *bn1b= (const float*)d_in[6];
  const float* c2W = (const float*)d_in[7],  *c2M = (const float*)d_in[8],  *c2G = (const float*)d_in[9];
  const float* bn2g= (const float*)d_in[10], *bn2b= (const float*)d_in[11];
  const float* c3W = (const float*)d_in[12], *c3M = (const float*)d_in[13], *c3G = (const float*)d_in[14];
  const float* bn3g= (const float*)d_in[15], *bn3b= (const float*)d_in[16];

  GruW gw; GruB gb;
  for (int L=0; L<6; ++L){
    int base = 17 + L*6;
    gw.w[L*3+0] = (const float*)d_in[base+0];   // wr
    gb.br[L]    = (const float*)d_in[base+1];
    gw.w[L*3+1] = (const float*)d_in[base+2];   // wu
    gb.bu[L]    = (const float*)d_in[base+3];
    gw.w[L*3+2] = (const float*)d_in[base+4];   // wc
    gb.bc[L]    = (const float*)d_in[base+5];
  }
  const float* c4W = (const float*)d_in[53], *c4M = (const float*)d_in[54], *c4G = (const float*)d_in[55];
  const float* bn4g= (const float*)d_in[56], *bn4b= (const float*)d_in[57];
  const float* wout= (const float*)d_in[58], *bout= (const float*)d_in[59];

  float*     wsf = (float*)d_ws;
  _Float16*  Ap  = (_Float16*)(wsf + APACK_F);
  float*     out = (float*)d_out;

  // weight prep
  k_pack_wm <<<(9975+255)/256, 256, 0, stream>>>(c1W,c1M,c2W,c2M,c3W,c3M,c4W,c4M, wsf);
  k_pack_gru<<<864, 128, 0, stream>>>(gw, Ap);

  // step1 featurizer with per-timestep batch statistics (deterministic)
  k_bn1_stats<<<NS, 128, 0, stream>>>(x, wsf, c1G);
  k_finalize <<<2, 256, 0, stream>>>(wsf, P1S_OFF, P1Q_OFF, MU1_OFF, RS1_OFF, 16, T_SEQ);
  k_bn2_stats<<<NS, 128, 0, stream>>>(x, wsf, c1G, c2G, bn1g, bn1b);
  k_finalize <<<4, 256, 0, stream>>>(wsf, P2S_OFF, P2Q_OFF, MU2_OFF, RS2_OFF, 48, T_SEQ);
  k_bn3_x4   <<<NS, 128, 0, stream>>>(x, timing, wsf, c1G, c2G, c3G, bn1g, bn1b, bn2g, bn2b);
  k_finalize <<<1, 64, 0, stream>>>(wsf, P3S_OFF, P3Q_OFF, MU3_OFF, RS3_OFF, 2, T_SEQ);

  // ConvGRU scan on WMMA, persistent per-sample workgroups
  // LDS: hb + act + ug + 2x bstage (65536 halves) + tap table = 140,936 B
  size_t lds = (size_t)(6*HID*NP + 96*NP + HID*NP + 2*NP*BSTR) * sizeof(_Float16)
             + (size_t)NTAP * sizeof(uint32_t);
  k_gru<<<BATCH, 256, lds, stream>>>(wsf, Ap, timing, bn3g, bn3b, gb);

  // head
  k_conv4   <<<BATCH, 128, 0, stream>>>(wsf, c4G);
  k_finalize<<<1, 64, 0, stream>>>(wsf, P4S_OFF, P4Q_OFF, MU4_OFF, RS4_OFF, 12, 1);
  k_out     <<<(BATCH*NPIX+255)/256, 256, 0, stream>>>(wsf, wout, bout, bn4g, bn4b, out);
}
